// GCN_EdgePool_41455024341479
// MI455X (gfx1250) — matrix operand order repaired and checked by reading, not compiled, not run on hardware
//
#include <hip/hip_runtime.h>
#include <hip/hip_bf16.h>
#include <math.h>

// ---------------- problem constants ----------------
#define N_NODES   60000
#define N_EDGES   600000
#define N_GRAPHS  8
#define F_IN      384
#define H_DIM     6
#define C_DIM     10
#define HS        8          // padded stride for hidden features
#define CS        16         // padded stride for class features
#define NPAD      (1<<20)    // bitonic sort size (>= N_EDGES)
#define CDIV(a,b) (((a)+(b)-1)/(b))

typedef __attribute__((ext_vector_type(16))) __bf16 v16bf;
typedef __attribute__((ext_vector_type(8)))  float  v8f;
typedef unsigned char      u8;
typedef unsigned int       u32;
typedef unsigned long long u64;

// monotone encode of float -> u32 (ascending order preserved)
__device__ __host__ inline u32 fenc_h(u32 u){ return (u & 0x80000000u) ? ~u : (u | 0x80000000u); }
__device__ inline u32 fenc(float f){ u32 u = __float_as_uint(f); return (u & 0x80000000u) ? ~u : (u | 0x80000000u); }
__device__ inline float fdec(u32 e){ u32 u = (e & 0x80000000u) ? (e & 0x7fffffffu) : ~e; return __uint_as_float(u); }

// ---------------- generic fills ----------------
__global__ void k_fill_f32(float* p, float v, int n){ int i=blockIdx.x*blockDim.x+threadIdx.x; if(i<n) p[i]=v; }
__global__ void k_fill_u32(u32* p, u32 v, int n){ int i=blockIdx.x*blockDim.x+threadIdx.x; if(i<n) p[i]=v; }
__global__ void k_fill_i32(int* p, int v, int n){ int i=blockIdx.x*blockDim.x+threadIdx.x; if(i<n) p[i]=v; }
__global__ void k_fill_u8 (u8* p, u8 v, int n){ int i=blockIdx.x*blockDim.x+threadIdx.x; if(i<n) p[i]=v; }
__global__ void k_fill_u64(u64* p, u64 v, int n){ int i=blockIdx.x*blockDim.x+threadIdx.x; if(i<n) p[i]=v; }

// ---------------- layer 1 projection: X(60000x384) @ W1(384x6) via WMMA bf16 ----------------
// pack W1 into per-lane fragment order: Wbp[(t*32 + lane)*16 + j] = W1pad[K = t*32 + 16*(lane>>4) + j][n = lane&15]
// so each lane's 32-byte B fragment is one aligned v16bf load.
__global__ void k_pack_w1(const float* __restrict__ W1, __bf16* __restrict__ Wbp){
  int i = blockIdx.x*blockDim.x + threadIdx.x;           // (F_IN/32)*32*16 = 6144
  if (i >= (F_IN/32)*32*16) return;
  int j = i & 15;
  int l = (i >> 4) & 31;
  int t = i >> 9;
  int K = t*32 + 16*(l >> 4) + j;
  int n = l & 15;
  Wbp[i] = (n < H_DIM) ? (__bf16)W1[K*H_DIM + n] : (__bf16)0.0f;
}

// one wave per 16-row tile, 8 waves per block (wave32). K loop: 384/32 = 12 WMMAs (fully unrolled).
// 60000 % 16 == 0, so every active tile is full: no row guards, EXEC stays all-ones for WMMA.
__global__ __launch_bounds__(256) void k_gemm1_wmma(const float* __restrict__ X,
                                                    const __bf16* __restrict__ Wbp,
                                                    float* __restrict__ P){
  int lane = threadIdx.x & 31;
  int wave = threadIdx.x >> 5;
  int mbase = (blockIdx.x*8 + wave) * 16;
  if (mbase >= N_NODES) return;                 // wave-uniform exit
  int half = lane >> 4;
  int n    = lane & 15;
  int mrow = mbase + n;
  const float* xr = X + (size_t)mrow * F_IN;

  v8f c = {};
  #pragma unroll
  for (int kb = 0; kb < F_IN; kb += 32) {
    // A fragment (16x32 bf16): per lane one contiguous run of 16 f32 (k = 16*half + s) -> 4x b128 loads
    const float4* p0 = (const float4*)(xr + kb + 16*half);
    const float4* p1 = (const float4*)(xr + kb + 16*half + 8);
    float4 q0 = p0[0], q1 = p0[1];
    float4 q2 = p1[0], q3 = p1[1];
    v16bf a;
    a[0]=(__bf16)q0.x; a[1]=(__bf16)q0.y; a[2]=(__bf16)q0.z; a[3]=(__bf16)q0.w;
    a[4]=(__bf16)q1.x; a[5]=(__bf16)q1.y; a[6]=(__bf16)q1.z; a[7]=(__bf16)q1.w;
    a[8]=(__bf16)q2.x; a[9]=(__bf16)q2.y; a[10]=(__bf16)q2.z; a[11]=(__bf16)q2.w;
    a[12]=(__bf16)q3.x; a[13]=(__bf16)q3.y; a[14]=(__bf16)q3.z; a[15]=(__bf16)q3.w;
    // B fragment: single aligned 32B load from the pre-packed panel
    v16bf b = *(const v16bf*)(Wbp + ((size_t)((kb >> 5)*32 + lane) << 4));
    c = __builtin_amdgcn_wmma_f32_16x16x32_bf16(false, a, false, b, (short)0, c, false, false);
  }
  // C/D layout: VGPR r, lanes 0-15 -> M=r; lanes 16-31 -> M=8+r; col = lane&15
  if (n < HS) {
    float* pr = P + (size_t)(mbase + 8*half)*HS + n;
    #pragma unroll
    for (int r = 0; r < 8; ++r) pr[(size_t)r*HS] = c[r];
  }
}

// ---------------- small projection for layer 2: h(Nx6) @ W2(6x10) ----------------
__global__ void k_proj2(const float* __restrict__ h, const float* __restrict__ W2,
                        float* __restrict__ P, int n){
  int i = blockIdx.x*blockDim.x + threadIdx.x;
  if (i >= n) return;
  float xv[H_DIM];
  #pragma unroll
  for (int j = 0; j < H_DIM; ++j) xv[j] = h[(size_t)i*HS + j];
  #pragma unroll
  for (int c = 0; c < CS; ++c) {
    float acc = 0.f;
    if (c < C_DIM) {
      #pragma unroll
      for (int j = 0; j < H_DIM; ++j) acc += xv[j] * W2[j*C_DIM + c];
    }
    P[(size_t)i*CS + c] = acc;
  }
}

// ---------------- GCNConv helpers ----------------
__global__ void k_deg(const int* __restrict__ src, const int* __restrict__ dst,
                      const u8* __restrict__ ev, float* __restrict__ degf,
                      u8* __restrict__ hasloop, int E){
  int e = blockIdx.x*blockDim.x + threadIdx.x;
  if (e >= E || !ev[e]) return;
  atomicAdd(&degf[dst[e]], 1.0f);
  if (src[e] == dst[e]) hasloop[dst[e]] = 1;
}

__global__ void k_dinv(const u8* __restrict__ nv, const u8* __restrict__ hasloop,
                       const float* __restrict__ degf, float* __restrict__ dinv,
                       u8* __restrict__ addl, int n){
  int i = blockIdx.x*blockDim.x + threadIdx.x;
  if (i >= n) return;
  u8 al = (nv[i] && !hasloop[i]) ? 1 : 0;
  addl[i] = al;
  float deg = degf[i] + (float)al;
  dinv[i] = (deg > 0.f) ? rsqrtf(fmaxf(deg, 1e-12f)) : 0.f;
}

__global__ void k_aggr(const int* __restrict__ src, const int* __restrict__ dst,
                       const u8* __restrict__ ev, const float* __restrict__ dinv,
                       const float* __restrict__ P, float* __restrict__ O,
                       int E, int FS, int FO){
  int e = blockIdx.x*blockDim.x + threadIdx.x;
  if (e >= E || !ev[e]) return;
  int s = src[e], d = dst[e];
  float coef = dinv[s] * dinv[d];
  const float* ps = P + (size_t)s*FS;
  float* od = O + (size_t)d*FS;
  for (int f = 0; f < FO; ++f) atomicAdd(&od[f], ps[f] * coef);
}

__global__ void k_gcn_fin(const u8* __restrict__ addl, const float* __restrict__ dinv,
                          const float* __restrict__ P, const float* __restrict__ bias,
                          float* __restrict__ O, int n, int FS, int FO, int relu){
  int i = blockIdx.x*blockDim.x + threadIdx.x;
  if (i >= n) return;
  float sl = addl[i] ? dinv[i]*dinv[i] : 0.f;
  for (int f = 0; f < FS; ++f) {
    float v = 0.f;
    if (f < FO) {
      v = O[(size_t)i*FS + f] + sl * P[(size_t)i*FS + f] + bias[f];
      if (relu) v = fmaxf(v, 0.f);
    }
    O[(size_t)i*FS + f] = v;
  }
}

// ---------------- EdgePooling ----------------
__global__ void k_raw(const float* __restrict__ x, const int* __restrict__ src,
                      const int* __restrict__ dst, const u8* __restrict__ ev,
                      const float* __restrict__ Wp, const float* __restrict__ bp,
                      float* __restrict__ raw, u32* __restrict__ menc,
                      int E, int FS, int FD){
  int e = blockIdx.x*blockDim.x + threadIdx.x;
  if (e >= E) return;
  if (!ev[e]) { raw[e] = 0.f; return; }
  const float* xs = x + (size_t)src[e]*FS;
  const float* xd = x + (size_t)dst[e]*FS;
  float r = bp[0];
  for (int f = 0; f < FD; ++f) r += xs[f]*Wp[f] + xd[f]*Wp[FD+f];
  raw[e] = r;
  atomicMax(&menc[dst[e]], fenc(r));
}

__global__ void k_ex(const float* __restrict__ raw, const int* __restrict__ dst,
                     const u8* __restrict__ ev, const u32* __restrict__ menc,
                     float* __restrict__ exv, float* __restrict__ sden, int E){
  int e = blockIdx.x*blockDim.x + threadIdx.x;
  if (e >= E) return;
  if (!ev[e]) { exv[e] = 0.f; return; }
  float m = fdec(menc[dst[e]]);
  if (!isfinite(m)) m = 0.f;
  float v = expf(raw[e] - m);
  exv[e] = v;
  atomicAdd(&sden[dst[e]], v);
}

// score + sort key: hi 32 bits = ~fenc(score) (descending), low 20 bits = edge index (stable)
__global__ void k_score(const float* __restrict__ exv, const int* __restrict__ dst,
                        const u8* __restrict__ ev, const float* __restrict__ sden,
                        float* __restrict__ score, u64* __restrict__ keys, int E){
  int e = blockIdx.x*blockDim.x + threadIdx.x;
  if (e >= E) return;
  float sc; u64 key;
  if (ev[e]) {
    float s = sden[dst[e]];
    sc = exv[e] / ((s > 0.f) ? s : 1.f) + 0.5f;
    key = ((u64)(~fenc(sc)) << 20) | (u32)e;
  } else {
    sc = 0.5f;
    key = (0xFFFFFFFFull << 20) | (u32)e;   // invalid -> last
  }
  score[e] = sc;
  keys[e]  = key;
}

__global__ void k_bitonic(u64* __restrict__ keys, int j, int k){
  int i = blockIdx.x*blockDim.x + threadIdx.x;
  if (i >= NPAD) return;
  int l = i ^ j;
  if (l > i) {
    u64 a = keys[i], b = keys[l];
    bool up = ((i & k) == 0);
    if (up ? (a > b) : (a < b)) { keys[i] = b; keys[l] = a; }
  }
}

// exact sequential greedy matching in sorted-score order (single lane)
__global__ void k_greedy(const u64* __restrict__ keys, const int* __restrict__ src,
                         const int* __restrict__ dst, const u8* __restrict__ ev,
                         u8* __restrict__ rem, u8* __restrict__ chosen, int E){
  if (blockIdx.x != 0 || threadIdx.x != 0) return;
  for (int t = 0; t < NPAD; ++t) {
    u32 e = (u32)(keys[t] & 0xFFFFFu);
    if (e >= (u32)E) continue;
    if (!ev[e]) continue;
    int s = src[e], d = dst[e];
    if (rem[s] && rem[d]) { chosen[e] = 1; rem[s] = 0; rem[d] = 0; }
  }
}

__global__ void k_dead(const int* __restrict__ src, const int* __restrict__ dst,
                       const u8* __restrict__ chosen, u8* __restrict__ deadf, int E){
  int e = blockIdx.x*blockDim.x + threadIdx.x;
  if (e >= E) return;
  if (chosen[e] && src[e] != dst[e]) deadf[dst[e]] = 1;
}

__global__ void k_newvalid(const u8* __restrict__ nv, const u8* __restrict__ deadf,
                           u8* __restrict__ nvo, int n){
  int i = blockIdx.x*blockDim.x + threadIdx.x;
  if (i >= n) return;
  nvo[i] = (nv[i] && !deadf[i]) ? 1 : 0;
}

__global__ void k_merge_acc(const int* __restrict__ src, const int* __restrict__ dst,
                            const u8* __restrict__ chosen, const float* __restrict__ score,
                            const float* __restrict__ x, float* __restrict__ addbuf,
                            float* __restrict__ scale, int E, int FS, int FD){
  int e = blockIdx.x*blockDim.x + threadIdx.x;
  if (e >= E || !chosen[e]) return;
  int s = src[e], d = dst[e];
  atomicAdd(&scale[s], score[e] - 1.f);
  if (s != d) {
    const float* xd = x + (size_t)d*FS;
    float* as = addbuf + (size_t)s*FS;
    for (int f = 0; f < FD; ++f) atomicAdd(&as[f], xd[f]);
  }
}

__global__ void k_newx(const float* __restrict__ x, const float* __restrict__ addbuf,
                       const float* __restrict__ scale, const u8* __restrict__ nvo,
                       float* __restrict__ xo, int n, int FS){
  int idx = blockIdx.x*blockDim.x + threadIdx.x;
  if (idx >= n*FS) return;
  int i = idx / FS;
  xo[idx] = nvo[i] ? (x[idx] + addbuf[idx]) * scale[i] : 0.f;
}

__global__ void k_msrc(const int* __restrict__ src, const int* __restrict__ dst,
                       const u8* __restrict__ chosen, int* __restrict__ msrc, int E){
  int e = blockIdx.x*blockDim.x + threadIdx.x;
  if (e >= E || !chosen[e]) return;
  atomicMax(&msrc[dst[e]], src[e]);
}

__global__ void k_cmap(const int* __restrict__ msrc, int* __restrict__ cmap, int n){
  int i = blockIdx.x*blockDim.x + threadIdx.x;
  if (i >= n) return;
  cmap[i] = (msrc[i] >= 0) ? msrc[i] : i;
}

// remap edges + build lexsort key: (ns_key:17 | nd:17 | idx:20) -- stable lexsort((nd, ns_k))
__global__ void k_remap(const int* __restrict__ src, const int* __restrict__ dst,
                        const u8* __restrict__ ev, const int* __restrict__ cmap,
                        int* __restrict__ ns, int* __restrict__ nd,
                        u64* __restrict__ keys, int E){
  int e = blockIdx.x*blockDim.x + threadIdx.x;
  if (e >= E) return;
  int a = cmap[src[e]], b = cmap[dst[e]];
  ns[e] = a; nd[e] = b;
  u64 ak = ev[e] ? (u64)(u32)a : (u64)N_NODES;
  keys[e] = (ak << 37) | ((u64)(u32)b << 20) | (u32)e;
}

__global__ void k_dedup(const u64* __restrict__ keys, const u8* __restrict__ ev,
                        u8* __restrict__ evo, int E){
  int t = blockIdx.x*blockDim.x + threadIdx.x;
  if (t >= NPAD) return;
  u64 k = keys[t];
  u32 e = (u32)(k & 0xFFFFFu);
  if (e >= (u32)E) return;
  bool first = (t == 0) || ((keys[t-1] >> 20) != (k >> 20));
  evo[e] = (first && ev[e]) ? 1 : 0;
}

// ---------------- readout ----------------
__global__ void k_pool_graph(const float* __restrict__ h, const int* __restrict__ batch,
                             const u8* __restrict__ nv, float* __restrict__ cnt,
                             float* __restrict__ sm, int n){
  int i = blockIdx.x*blockDim.x + threadIdx.x;
  if (i >= n || !nv[i]) return;
  int b = batch[i];
  atomicAdd(&cnt[b], 1.f);
  for (int f = 0; f < C_DIM; ++f) atomicAdd(&sm[b*CS + f], h[(size_t)i*CS + f]);
}

__global__ void k_logsoftmax(const float* __restrict__ cnt, const float* __restrict__ sm,
                             float* __restrict__ out){
  int i = threadIdx.x;
  if (i >= N_GRAPHS*C_DIM) return;
  int b = i / C_DIM, c = i % C_DIM;
  float cn = cnt[b];
  float g[C_DIM], mx = -INFINITY;
  #pragma unroll
  for (int j = 0; j < C_DIM; ++j) { g[j] = sm[b*CS + j] / cn; mx = fmaxf(mx, g[j]); }
  float se = 0.f;
  #pragma unroll
  for (int j = 0; j < C_DIM; ++j) se += expf(g[j] - mx);
  out[i] = g[c] - mx - logf(se);
}

// ---------------- host orchestration ----------------
static void run_bitonic(hipStream_t st, u64* keys){
  for (int k = 2; k <= NPAD; k <<= 1)
    for (int j = k >> 1; j > 0; j >>= 1)
      k_bitonic<<<CDIV(NPAD,256),256,0,st>>>(keys, j, k);
}

static void run_gcn(hipStream_t st, const float* proj, const int* src, const int* dst,
                    const u8* ev, const u8* nv, const float* bias, float* out,
                    int FS, int FO, int relu,
                    float* degf, u8* hasloop, float* dinv, u8* addl){
  k_fill_f32<<<CDIV(N_NODES,256),256,0,st>>>(degf, 0.f, N_NODES);
  k_fill_u8 <<<CDIV(N_NODES,256),256,0,st>>>(hasloop, 0, N_NODES);
  k_deg<<<CDIV(N_EDGES,256),256,0,st>>>(src, dst, ev, degf, hasloop, N_EDGES);
  k_dinv<<<CDIV(N_NODES,256),256,0,st>>>(nv, hasloop, degf, dinv, addl, N_NODES);
  k_fill_f32<<<CDIV(N_NODES*FS,256),256,0,st>>>(out, 0.f, N_NODES*FS);
  k_aggr<<<CDIV(N_EDGES,256),256,0,st>>>(src, dst, ev, dinv, proj, out, N_EDGES, FS, FO);
  k_gcn_fin<<<CDIV(N_NODES,256),256,0,st>>>(addl, dinv, proj, bias, out, N_NODES, FS, FO, relu);
}

static void run_edge_pool(hipStream_t st, const float* x, const int* src, const int* dst,
                          const u8* ev, const u8* nv, const float* Wp, const float* bp,
                          int FS, int FD,
                          float* raw, u32* menc, float* exv, float* sden, float* score,
                          u64* keys, u8* chosen, u8* rem, u8* deadf, int* msrc, int* cmap,
                          float* addbuf, float* scale,
                          float* xo, int* ns, int* nd, u8* evo, u8* nvo){
  const u32 ENC_NEG_INF = fenc_h(0xFF800000u); // fenc(-inf)
  k_fill_u32<<<CDIV(N_NODES,256),256,0,st>>>(menc, ENC_NEG_INF, N_NODES);
  k_fill_f32<<<CDIV(N_NODES,256),256,0,st>>>(sden, 0.f, N_NODES);
  k_raw  <<<CDIV(N_EDGES,256),256,0,st>>>(x, src, dst, ev, Wp, bp, raw, menc, N_EDGES, FS, FD);
  k_ex   <<<CDIV(N_EDGES,256),256,0,st>>>(raw, dst, ev, menc, exv, sden, N_EDGES);
  k_score<<<CDIV(N_EDGES,256),256,0,st>>>(exv, dst, ev, sden, score, keys, N_EDGES);
  k_fill_u64<<<CDIV(NPAD-N_EDGES,256),256,0,st>>>(keys + N_EDGES, ~0ull, NPAD - N_EDGES);
  run_bitonic(st, keys);
  (void)hipMemcpyAsync(rem, nv, N_NODES, hipMemcpyDeviceToDevice, st);
  k_fill_u8<<<CDIV(N_EDGES,256),256,0,st>>>(chosen, 0, N_EDGES);
  k_greedy<<<1,1,0,st>>>(keys, src, dst, ev, rem, chosen, N_EDGES);
  k_fill_u8<<<CDIV(N_NODES,256),256,0,st>>>(deadf, 0, N_NODES);
  k_dead<<<CDIV(N_EDGES,256),256,0,st>>>(src, dst, chosen, deadf, N_EDGES);
  k_newvalid<<<CDIV(N_NODES,256),256,0,st>>>(nv, deadf, nvo, N_NODES);
  k_fill_f32<<<CDIV(N_NODES*FS,256),256,0,st>>>(addbuf, 0.f, N_NODES*FS);
  k_fill_f32<<<CDIV(N_NODES,256),256,0,st>>>(scale, 1.f, N_NODES);
  k_merge_acc<<<CDIV(N_EDGES,256),256,0,st>>>(src, dst, chosen, score, x, addbuf, scale, N_EDGES, FS, FD);
  k_newx<<<CDIV(N_NODES*FS,256),256,0,st>>>(x, addbuf, scale, nvo, xo, N_NODES, FS);
  k_fill_i32<<<CDIV(N_NODES,256),256,0,st>>>(msrc, -1, N_NODES);
  k_msrc<<<CDIV(N_EDGES,256),256,0,st>>>(src, dst, chosen, msrc, N_EDGES);
  k_cmap<<<CDIV(N_NODES,256),256,0,st>>>(msrc, cmap, N_NODES);
  k_remap<<<CDIV(N_EDGES,256),256,0,st>>>(src, dst, ev, cmap, ns, nd, keys, N_EDGES);
  k_fill_u64<<<CDIV(NPAD-N_EDGES,256),256,0,st>>>(keys + N_EDGES, ~0ull, NPAD - N_EDGES);
  run_bitonic(st, keys);
  k_fill_u8<<<CDIV(N_EDGES,256),256,0,st>>>(evo, 0, N_EDGES);
  k_dedup<<<CDIV(NPAD,256),256,0,st>>>(keys, ev, evo, N_EDGES);
}

extern "C" void kernel_launch(void* const* d_in, const int* in_sizes, int n_in,
                              void* d_out, int out_size, void* d_ws, size_t ws_size,
                              hipStream_t stream) {
  (void)in_sizes; (void)n_in; (void)out_size; (void)ws_size;
  const float* x   = (const float*)d_in[0];
  const int*   src = (const int*)d_in[1];
  const int*   dst = (const int*)d_in[2];
  const int*   bat = (const int*)d_in[3];
  const float* W1  = (const float*)d_in[4];
  const float* b1  = (const float*)d_in[5];
  const float* W2  = (const float*)d_in[6];
  const float* b2  = (const float*)d_in[7];
  const float* Wp1 = (const float*)d_in[8];
  const float* bp1 = (const float*)d_in[9];
  const float* Wp2 = (const float*)d_in[10];
  const float* bp2 = (const float*)d_in[11];
  float* out = (float*)d_out;

  char* base = (char*)d_ws;
  size_t off = 0;
  auto alloc = [&](size_t bytes) -> void* {
    void* p = base + off;
    off = (off + bytes + 255) & ~(size_t)255;
    return p;
  };

  __bf16* w1bf   = (__bf16*)alloc((size_t)F_IN*16*sizeof(__bf16));
  float* proj0   = (float*)alloc((size_t)N_NODES*HS*sizeof(float));
  float* h1      = (float*)alloc((size_t)N_NODES*HS*sizeof(float));
  float* hp1     = (float*)alloc((size_t)N_NODES*HS*sizeof(float));
  float* proj1   = (float*)alloc((size_t)N_NODES*CS*sizeof(float));
  float* h2      = (float*)alloc((size_t)N_NODES*CS*sizeof(float));
  float* hp2     = (float*)alloc((size_t)N_NODES*CS*sizeof(float));
  float* degf    = (float*)alloc((size_t)N_NODES*sizeof(float));
  float* dinv    = (float*)alloc((size_t)N_NODES*sizeof(float));
  float* raw     = (float*)alloc((size_t)N_EDGES*sizeof(float));
  float* sden    = (float*)alloc((size_t)N_NODES*sizeof(float));
  float* exv     = (float*)alloc((size_t)N_EDGES*sizeof(float));
  float* score   = (float*)alloc((size_t)N_EDGES*sizeof(float));
  float* addbuf  = (float*)alloc((size_t)N_NODES*CS*sizeof(float));
  float* scale   = (float*)alloc((size_t)N_NODES*sizeof(float));
  float* cnt     = (float*)alloc((size_t)N_GRAPHS*sizeof(float));
  float* smb     = (float*)alloc((size_t)N_GRAPHS*CS*sizeof(float));
  u32*   menc    = (u32*)alloc((size_t)N_NODES*sizeof(u32));
  u64*   keys    = (u64*)alloc((size_t)NPAD*sizeof(u64));
  int*   s1      = (int*)alloc((size_t)N_EDGES*sizeof(int));
  int*   d1      = (int*)alloc((size_t)N_EDGES*sizeof(int));
  int*   s2      = (int*)alloc((size_t)N_EDGES*sizeof(int));
  int*   d2      = (int*)alloc((size_t)N_EDGES*sizeof(int));
  int*   msrc    = (int*)alloc((size_t)N_NODES*sizeof(int));
  int*   cmap    = (int*)alloc((size_t)N_NODES*sizeof(int));
  u8*    ev0     = (u8*)alloc((size_t)N_EDGES);
  u8*    ev1     = (u8*)alloc((size_t)N_EDGES);
  u8*    ev2     = (u8*)alloc((size_t)N_EDGES);
  u8*    chosen  = (u8*)alloc((size_t)N_EDGES);
  u8*    nv0     = (u8*)alloc((size_t)N_NODES);
  u8*    nv1     = (u8*)alloc((size_t)N_NODES);
  u8*    nv2     = (u8*)alloc((size_t)N_NODES);
  u8*    hasloop = (u8*)alloc((size_t)N_NODES);
  u8*    addl    = (u8*)alloc((size_t)N_NODES);
  u8*    rem     = (u8*)alloc((size_t)N_NODES);
  u8*    deadf   = (u8*)alloc((size_t)N_NODES);

  // initial validity masks (all valid)
  k_fill_u8<<<CDIV(N_EDGES,256),256,0,stream>>>(ev0, 1, N_EDGES);
  k_fill_u8<<<CDIV(N_NODES,256),256,0,stream>>>(nv0, 1, N_NODES);

  // ---- GCNConv 1: WMMA projection + normalized aggregation + ReLU ----
  k_pack_w1<<<CDIV(F_IN*16,256),256,0,stream>>>(W1, w1bf);
  k_gemm1_wmma<<<CDIV(N_NODES,128),256,0,stream>>>(x, w1bf, proj0);
  run_gcn(stream, proj0, src, dst, ev0, nv0, b1, h1, HS, H_DIM, /*relu=*/1,
          degf, hasloop, dinv, addl);

  // ---- EdgePooling 1 ----
  run_edge_pool(stream, h1, src, dst, ev0, nv0, Wp1, bp1, HS, H_DIM,
                raw, menc, exv, sden, score, keys, chosen, rem, deadf, msrc, cmap,
                addbuf, scale, hp1, s1, d1, ev1, nv1);

  // ---- GCNConv 2 ----
  k_proj2<<<CDIV(N_NODES,256),256,0,stream>>>(hp1, W2, proj1, N_NODES);
  run_gcn(stream, proj1, s1, d1, ev1, nv1, b2, h2, CS, C_DIM, /*relu=*/0,
          degf, hasloop, dinv, addl);

  // ---- EdgePooling 2 ----
  run_edge_pool(stream, h2, s1, d1, ev1, nv1, Wp2, bp2, CS, C_DIM,
                raw, menc, exv, sden, score, keys, chosen, rem, deadf, msrc, cmap,
                addbuf, scale, hp2, s2, d2, ev2, nv2);

  // ---- masked mean pool per graph + log_softmax ----
  k_fill_f32<<<1,64,0,stream>>>(cnt, 0.f, N_GRAPHS);
  k_fill_f32<<<1,256,0,stream>>>(smb, 0.f, N_GRAPHS*CS);
  k_pool_graph<<<CDIV(N_NODES,256),256,0,stream>>>(hp2, bat, nv2, cnt, smb, N_NODES);
  k_logsoftmax<<<1,128,0,stream>>>(cnt, smb, out);
}
